// Gaussian_25744033972454
// MI455X (gfx1250) — matrix operand order repaired and checked by reading, not compiled, not run on hardware
//
#include <hip/hip_runtime.h>

typedef __bf16 bf16_t;
typedef __attribute__((ext_vector_type(16))) __bf16 v16bf;
typedef __attribute__((ext_vector_type(8)))  float  v8f;

// pointer-parameter types for the async global->LDS builtins
typedef int v4i_vs __attribute__((vector_size(16)));
typedef __attribute__((address_space(1))) v4i_vs v4i_as1;   // global
typedef __attribute__((address_space(3))) v4i_vs v4i_as3;   // LDS

#define TILE_M 128
#define TILE_N 128
#define TILE_K 32
#define LDSK   40   // padded K row (40 bf16 = 80B) -> conflict-free b128 frag reads

// ---------------------------------------------------------------------------
// CDNA5 async global->LDS copy (16B per lane), ASYNCcnt-tracked
// ---------------------------------------------------------------------------
__device__ __forceinline__ void async_copy_b128(const bf16_t* g, bf16_t* l) {
#if __has_builtin(__builtin_amdgcn_global_load_async_to_lds_b128)
    __builtin_amdgcn_global_load_async_to_lds_b128(
        (v4i_as1*)(uintptr_t)g,
        (v4i_as3*)(unsigned)(uintptr_t)l,
        0, 0);
#else
    asm volatile("global_load_async_to_lds_b128 %0, %1, off"
                 :: "v"((unsigned)(uintptr_t)l), "v"(g) : "memory");
#endif
}

__device__ __forceinline__ void wait_async0() {
#if __has_builtin(__builtin_amdgcn_s_wait_asynccnt)
    __builtin_amdgcn_s_wait_asynccnt(0);
#else
    asm volatile("s_wait_asynccnt 0x0" ::: "memory");
#endif
}

// ---------------------------------------------------------------------------
// Sparse setup: degree count, t1m = (1-MU)*I - D^-1 W scatter
// ---------------------------------------------------------------------------
__global__ void deg_zero_kernel(float* __restrict__ deg, int N) {
    int i = blockIdx.x * blockDim.x + threadIdx.x;
    if (i < N) deg[i] = 0.0f;
}

__global__ void deg_count_kernel(const int* __restrict__ eidx, int E,
                                 float* __restrict__ deg) {
    int e = blockIdx.x * blockDim.x + threadIdx.x;
    if (e < E) atomicAdd(&deg[eidx[e]], 1.0f);
}

__global__ void t1m_init_kernel(float* __restrict__ t1m, int N, float dval) {
    size_t i = (size_t)blockIdx.x * blockDim.x + threadIdx.x;
    size_t r = i / (unsigned)N;
    size_t c = i - r * (unsigned)N;
    t1m[i] = (r == c) ? dval : 0.0f;
}

__global__ void edge_scatter_kernel(const int* __restrict__ eidx,
                                    const float* __restrict__ attr,
                                    const float* __restrict__ deg,
                                    float* __restrict__ t1m, int N, int E) {
    int e = blockIdx.x * blockDim.x + threadIdx.x;
    if (e < E) {
        int r = eidx[e];
        int c = eidx[E + e];
        atomicAdd(&t1m[(size_t)r * N + c], -attr[e] / deg[r]);
    }
}

// ---------------------------------------------------------------------------
// One-shot fp32 -> bf16 hi/lo split, row-major (for the A operand)
// ---------------------------------------------------------------------------
__global__ __launch_bounds__(256)
void split_rowmajor_kernel(const float* __restrict__ src,
                           bf16_t* __restrict__ hi, bf16_t* __restrict__ lo,
                           size_t n4) {
    size_t i = (size_t)blockIdx.x * blockDim.x + threadIdx.x;
    if (i >= n4) return;
    float4 v = ((const float4*)src)[i];
    union { bf16_t h[4]; uint2 u; } ph, pl;
    ph.h[0] = (bf16_t)v.x; pl.h[0] = (bf16_t)(v.x - (float)ph.h[0]);
    ph.h[1] = (bf16_t)v.y; pl.h[1] = (bf16_t)(v.y - (float)ph.h[1]);
    ph.h[2] = (bf16_t)v.z; pl.h[2] = (bf16_t)(v.z - (float)ph.h[2]);
    ph.h[3] = (bf16_t)v.w; pl.h[3] = (bf16_t)(v.w - (float)ph.h[3]);
    ((uint2*)hi)[i] = ph.u;
    ((uint2*)lo)[i] = pl.u;
}

// ---------------------------------------------------------------------------
// One-shot fp32 -> bf16 hi/lo split WITH transpose (for the B operand)
// ---------------------------------------------------------------------------
__global__ __launch_bounds__(256)
void split_transpose_kernel(const float* __restrict__ src,
                            bf16_t* __restrict__ hiT, bf16_t* __restrict__ loT,
                            int N) {
    __shared__ float tile[64][65];
    const int tx = threadIdx.x & 63;
    const int ty = threadIdx.x >> 6;          // 0..3
    const int rb = blockIdx.y * 64;
    const int cb = blockIdx.x * 64;
#pragma unroll
    for (int j = 0; j < 16; ++j) {
        int r = ty * 16 + j;
        tile[r][tx] = src[(size_t)(rb + r) * N + cb + tx];
    }
    __syncthreads();
#pragma unroll
    for (int j = 0; j < 16; ++j) {
        int c = ty * 16 + j;                  // source column = dest row
        float v = tile[tx][c];
        bf16_t h = (bf16_t)v;
        size_t idx = (size_t)(cb + c) * N + rb + tx;
        hiT[idx] = h;
        loT[idx] = (bf16_t)(v - (float)h);
    }
}

// ---------------------------------------------------------------------------
// Split-precision bf16x3 WMMA GEMM with fused Chebyshev epilogue:
//   out = alpha*(A@B) + ce1*e1 + ce2*e2 + cdiag*I
// Pre-split bf16 operands: A row-major hi/lo, B column-major hi/lo.
// Tile staging via GLOBAL_LOAD_ASYNC_TO_LDS_B128, double-buffered LDS,
// DMA of tile k+1 overlaps the WMMAs of tile k; one barrier per K-tile.
// ---------------------------------------------------------------------------
__global__ __launch_bounds__(256)
void cheb_gemm_kernel(float* __restrict__ out,
                      const bf16_t* __restrict__ Ahi, const bf16_t* __restrict__ Alo,
                      const bf16_t* __restrict__ BhiT, const bf16_t* __restrict__ BloT,
                      int N, float alpha,
                      const float* __restrict__ e1, float ce1,
                      const float* __restrict__ e2, float ce2,
                      float cdiag)
{
    __shared__ bf16_t Ah[2][TILE_M][LDSK];
    __shared__ bf16_t Al[2][TILE_M][LDSK];
    __shared__ bf16_t Bh[2][TILE_N][LDSK];   // [n][k]
    __shared__ bf16_t Bl[2][TILE_N][LDSK];

    const int tid  = threadIdx.x;
    const int lane = tid & 31;
    const int wave = tid >> 5;
    const int wm   = wave & 3;   // 4 waves along M: 32-row strips
    const int wn   = wave >> 2;  // 2 waves along N: 64-col strips
    const int half = lane >> 4;
    const int lr   = lane & 15;

    const int rowBase = blockIdx.y * TILE_M;
    const int colBase = blockIdx.x * TILE_N;

    // staging map: per matrix form, 512 x 16B (128 rows x 32 bf16); 2/thread
    const int sm0 = tid >> 2,         sq0 = tid & 3;
    const int sm1 = (256 + tid) >> 2, sq1 = (256 + tid) & 3;
    const size_t gA0 = (size_t)(rowBase + sm0) * N + sq0 * 8;
    const size_t gA1 = (size_t)(rowBase + sm1) * N + sq1 * 8;
    const size_t gB0 = (size_t)(colBase + sm0) * N + sq0 * 8;
    const size_t gB1 = (size_t)(colBase + sm1) * N + sq1 * 8;

    v8f acc[2][4];
#pragma unroll
    for (int t = 0; t < 2; ++t)
#pragma unroll
        for (int u = 0; u < 4; ++u)
            acc[t][u] = (v8f){0.f,0.f,0.f,0.f,0.f,0.f,0.f,0.f};

    // issue async DMA of one K-tile (4 operand forms) into LDS buffer p
    auto stage = [&](int p, int k0) {
        async_copy_b128(Ahi  + gA0 + k0, &Ah[p][sm0][sq0 * 8]);
        async_copy_b128(Ahi  + gA1 + k0, &Ah[p][sm1][sq1 * 8]);
        async_copy_b128(Alo  + gA0 + k0, &Al[p][sm0][sq0 * 8]);
        async_copy_b128(Alo  + gA1 + k0, &Al[p][sm1][sq1 * 8]);
        async_copy_b128(BhiT + gB0 + k0, &Bh[p][sm0][sq0 * 8]);
        async_copy_b128(BhiT + gB1 + k0, &Bh[p][sm1][sq1 * 8]);
        async_copy_b128(BloT + gB0 + k0, &Bl[p][sm0][sq0 * 8]);
        async_copy_b128(BloT + gB1 + k0, &Bl[p][sm1][sq1 * 8]);
    };

    const int nk = N / TILE_K;
    stage(0, 0);   // prologue DMA

    for (int kt = 0; kt < nk; ++kt) {
        const int p = kt & 1;

        wait_async0();     // my DMAs for tile kt landed in LDS
        __syncthreads();   // everyone's landed; all done reading buffer 1-p

        if (kt + 1 < nk)   // DMA for tile kt+1 overlaps the WMMAs below
            stage(1 - p, (kt + 1) * TILE_K);
        if (kt + 2 < nk) { // pull tile kt+2 toward L2 (global_prefetch_b8)
            __builtin_prefetch(Ahi  + gA0 + (kt + 2) * TILE_K, 0, 1);
            __builtin_prefetch(BhiT + gB0 + (kt + 2) * TILE_K, 0, 1);
        }

        // ---- fragment loads + 24 WMMAs per wave ----
        union FragU { uint4 q[2]; v16bf v; };
        // A frag: lane holds row M=lr, K in {kb..kb+7} U {kb+16..kb+23}, kb=8*half
        FragU ah[2], al[2];
#pragma unroll
        for (int t = 0; t < 2; ++t) {
            const uint4* ph = (const uint4*)&Ah[p][wm * 32 + t * 16 + lr][half * 8];
            const uint4* pl = (const uint4*)&Al[p][wm * 32 + t * 16 + lr][half * 8];
            ah[t].q[0] = ph[0]; ah[t].q[1] = ph[2];
            al[t].q[0] = pl[0]; al[t].q[1] = pl[2];
        }
#pragma unroll
        for (int u = 0; u < 4; ++u) {
            // B frag: lane holds column N=lr, K in {kb..kb+7} U {kb+16..kb+23}, kb=8*half
            FragU bh, bl;
            const uint4* ph = (const uint4*)&Bh[p][wn * 64 + u * 16 + lr][half * 8];
            const uint4* pl = (const uint4*)&Bl[p][wn * 64 + u * 16 + lr][half * 8];
            bh.q[0] = ph[0]; bh.q[1] = ph[2];
            bl.q[0] = pl[0]; bl.q[1] = pl[2];
#pragma unroll
            for (int t = 0; t < 2; ++t) {
                acc[t][u] = __builtin_amdgcn_wmma_f32_16x16x32_bf16(
                    false, ah[t].v, false, bh.v, (short)0, acc[t][u], false, false);
                acc[t][u] = __builtin_amdgcn_wmma_f32_16x16x32_bf16(
                    false, ah[t].v, false, bl.v, (short)0, acc[t][u], false, false);
                acc[t][u] = __builtin_amdgcn_wmma_f32_16x16x32_bf16(
                    false, al[t].v, false, bh.v, (short)0, acc[t][u], false, false);
            }
        }
    }

    // ---- fused epilogue ----
#pragma unroll
    for (int t = 0; t < 2; ++t) {
#pragma unroll
        for (int u = 0; u < 4; ++u) {
#pragma unroll
            for (int r = 0; r < 8; ++r) {
                int row = rowBase + wm * 32 + t * 16 + half * 8 + r;
                int col = colBase + wn * 64 + u * 16 + lr;
                size_t idx = (size_t)row * N + col;
                float v = alpha * acc[t][u][r];
                if (e1) v += ce1 * e1[idx];
                if (e2) v += ce2 * e2[idx];
                if (row == col) v += cdiag;
                out[idx] = v;
            }
        }
    }
}

// ---------------------------------------------------------------------------
extern "C" void kernel_launch(void* const* d_in, const int* in_sizes, int n_in,
                              void* d_out, int out_size, void* d_ws, size_t ws_size,
                              hipStream_t stream) {
    const int D = 128;
    const int N = in_sizes[0] / D;     // 4096
    const int E = in_sizes[2];         // 131072
    const size_t NN = (size_t)N * N;

    const int*   eidx = (const int*)d_in[1];
    const float* attr = (const float*)d_in[2];
    float*       out  = (float*)d_out;

    // workspace layout (floats): bufA | bufB | 4 bf16 forms (NN/2 floats each) | deg
    float*  bufA = (float*)d_ws;          // t1m, then T2 (fp32)
    float*  bufB = bufA + NN;             // lx (fp32)
    bf16_t* Whi  = (bf16_t*)(bufB + NN);  // A-operand hi (row-major)
    bf16_t* Wlo  = Whi + NN;
    bf16_t* WhiT = Wlo + NN;              // B-operand hi (col-major)
    bf16_t* WloT = WhiT + NN;
    float*  deg  = (float*)(WloT + NN);

    const float MU = 0.2f;
    const float C0 = 1.2660658777520082f;
    const float C1 = 1.1303182079849703f;
    const float C2 = -0.27149533953407656f;
    const float C3 = 0.044336849848663804f;

    deg_zero_kernel<<<(N + 255) / 256, 256, 0, stream>>>(deg, N);
    deg_count_kernel<<<(E + 255) / 256, 256, 0, stream>>>(eidx, E, deg);
    t1m_init_kernel<<<(int)(NN / 256), 256, 0, stream>>>(bufA, N, 1.0f - MU);
    edge_scatter_kernel<<<(E + 255) / 256, 256, 0, stream>>>(eidx, attr, deg, bufA, N, E);

    dim3 ggrid(N / TILE_N, N / TILE_M);
    dim3 tgrid(N / 64, N / 64);
    const int splitBlocks = (int)((NN / 4 + 255) / 256);

    // --- lx = -0.5*(t1m@t1m) + 0.5*I ---
    split_rowmajor_kernel<<<splitBlocks, 256, 0, stream>>>(bufA, Whi, Wlo, NN / 4);
    split_transpose_kernel<<<tgrid, 256, 0, stream>>>(bufA, WhiT, WloT, N);
    cheb_gemm_kernel<<<ggrid, 256, 0, stream>>>(bufB, Whi, Wlo, WhiT, WloT, N, -0.5f,
                                                nullptr, 0.f, nullptr, 0.f, 0.5f);

    // --- T2 = 2*(lx@lx) - I  (into bufA) ---
    split_rowmajor_kernel<<<splitBlocks, 256, 0, stream>>>(bufB, Whi, Wlo, NN / 4);
    split_transpose_kernel<<<tgrid, 256, 0, stream>>>(bufB, WhiT, WloT, N);
    cheb_gemm_kernel<<<ggrid, 256, 0, stream>>>(bufA, Whi, Wlo, WhiT, WloT, N, 2.0f,
                                                nullptr, 0.f, nullptr, 0.f, -1.0f);

    // --- res = 2*C3*(lx@T2) + (C1-C3)*lx + C2*T2 + C0*I ---
    // Whi/Wlo still hold lx row-major; refresh only the B (col-major) forms to T2.
    split_transpose_kernel<<<tgrid, 256, 0, stream>>>(bufA, WhiT, WloT, N);
    cheb_gemm_kernel<<<ggrid, 256, 0, stream>>>(out, Whi, Wlo, WhiT, WloT, N, 2.0f * C3,
                                                bufB, C1 - C3, bufA, C2, C0);
}
